// OrionBiAxialChannelAttention_8169027797328
// MI455X (gfx1250) — matrix operand order repaired and checked by reading, not compiled, not run on hardware
//
#include <hip/hip_runtime.h>

// ---------------------------------------------------------------------------
// Types
// ---------------------------------------------------------------------------
typedef __attribute__((ext_vector_type(16))) __bf16       v16bf;
typedef __attribute__((ext_vector_type(8)))  float        v8f;
typedef __attribute__((ext_vector_type(4)))  unsigned int u32x4;
typedef int v4i_gcc __attribute__((__vector_size__(16)));   // matches async-LDS builtin param
typedef unsigned short bf16_t;   // raw bf16 storage

static_assert(sizeof(v16bf) == 32, "v16bf must be 32 bytes");

#if defined(__gfx1250__) && \
    __has_builtin(__builtin_amdgcn_global_load_async_to_lds_b128) && \
    __has_builtin(__builtin_amdgcn_s_wait_asynccnt)
#define USE_ASYNC_LDS 1
#else
#define USE_ASYNC_LDS 0
#endif

#define GAS __attribute__((address_space(1)))
#define LAS __attribute__((address_space(3)))

__device__ __forceinline__ bf16_t f2bf(float f) {
    unsigned u = __float_as_uint(f);
    unsigned r = u + 0x7FFFu + ((u >> 16) & 1u);   // round-to-nearest-even
    return (bf16_t)(r >> 16);
}

// ---------------------------------------------------------------------------
// f32 -> bf16 convert (weights)
// ---------------------------------------------------------------------------
__global__ __launch_bounds__(256) void k_f2bf(const float* __restrict__ in,
                                              bf16_t* __restrict__ out, int n) {
    int i = blockIdx.x * 256 + threadIdx.x;
    if (i < n) out[i] = f2bf(in[i]);
}

// ---------------------------------------------------------------------------
// Row LayerNorm (biased var, eps=1e-5). One wave per row. outF/outB optional.
// ---------------------------------------------------------------------------
__global__ __launch_bounds__(256) void k_ln(const float* __restrict__ in,
                                            const float* __restrict__ g,
                                            const float* __restrict__ b,
                                            float* __restrict__ outF,
                                            bf16_t* __restrict__ outB,
                                            int rows, int D) {
    int wave = threadIdx.x >> 5, lane = threadIdx.x & 31;
    int row = blockIdx.x * 8 + wave;
    if (row >= rows) return;
    const float* p = in + (size_t)row * D;
    float s = 0.f, sq = 0.f;
    for (int i = lane; i < D; i += 32) { float v = p[i]; s += v; sq += v * v; }
    for (int o = 16; o; o >>= 1) { s += __shfl_xor(s, o, 32); sq += __shfl_xor(sq, o, 32); }
    float m = s / D;
    float var = sq / D - m * m;
    float r = rsqrtf(var + 1e-5f);
    for (int i = lane; i < D; i += 32) {
        float v = (p[i] - m) * r * g[i] + b[i];
        if (outF) outF[(size_t)row * D + i] = v;
        if (outB) outB[(size_t)row * D + i] = f2bf(v);
    }
}

// ---------------------------------------------------------------------------
// WMMA GEMM: C[M,N] = act(A[M,K] * W[N,K]^T + bias) (+ R).  A,W bf16; out f32/bf16.
// Workgroup tile 128x128, 8 waves as 4(M)x2(N), wave tile 32x64, K step 32.
// Async path: double-buffered LDS, GLOBAL_LOAD_ASYNC_TO_LDS_B128 of tile k+1
// overlapped with WMMA on tile k (ASYNCcnt + workgroup barrier).
// ---------------------------------------------------------------------------
#define LDSS 40   // LDS row stride in bf16 elems (32 data + 8 pad -> 80B, 16B aligned)
#if USE_ASYNC_LDS
#define NBUF 2
#else
#define NBUF 1
#endif
#define TILE_ELEMS (128 * LDSS)

__global__ __launch_bounds__(256) void k_gemm(const bf16_t* __restrict__ A,
                                              const bf16_t* __restrict__ W,
                                              const float* __restrict__ bias,
                                              const float* __restrict__ R,
                                              float* __restrict__ outF,
                                              bf16_t* __restrict__ outB,
                                              int M, int N, int K, int act) {
    __shared__ bf16_t lA[NBUF * TILE_ELEMS];
    __shared__ bf16_t lW[NBUF * TILE_ELEMS];
    const int m0 = blockIdx.y * 128, n0 = blockIdx.x * 128;
    const int tid = threadIdx.x, lane = tid & 31, wave = tid >> 5;
    const int wm = (wave & 3) * 32, wn = (wave >> 2) * 64;

    v8f acc[2][4];
#pragma unroll
    for (int i = 0; i < 2; i++)
#pragma unroll
        for (int j = 0; j < 4; j++)
#pragma unroll
            for (int e = 0; e < 8; e++) acc[i][j][e] = 0.f;

    // loaders: 128 rows x 4 x 16B chunks per matrix; each thread moves 2 chunks per matrix
    const int rl0 = tid >> 2, rl1 = 64 + (tid >> 2), sg = tid & 3;

    // fragment consumption from LDS buffer `buf`, 8 WMMAs
    auto compute_tile = [&](int buf) {
        const bf16_t* bA = lA + buf * TILE_ELEMS;
        const bf16_t* bW = lW + buf * TILE_ELEMS;
        union F { u32x4 q[2]; v16bf v; };
        F a[2], bfr[4];
        const int l16 = lane & 15, hi = lane >> 4;
        // A frag (16x32): lanes 0-15 row m, K {0..7}U{16..23}; lanes 16-31 K {8..15}U{24..31}
#pragma unroll
        for (int i = 0; i < 2; i++) {
            const bf16_t* base = bA + (wm + i * 16 + l16) * LDSS;
            a[i].q[0] = *(const u32x4*)(base + hi * 8);
            a[i].q[1] = *(const u32x4*)(base + 16 + hi * 8);
        }
        // B frag (32x16): lanes 0-15 col n, K {0..7}U{16..23}; lanes 16-31 K {8..15}U{24..31}
#pragma unroll
        for (int j = 0; j < 4; j++) {
            const bf16_t* base = bW + (wn + j * 16 + l16) * LDSS;
            bfr[j].q[0] = *(const u32x4*)(base + hi * 8);
            bfr[j].q[1] = *(const u32x4*)(base + 16 + hi * 8);
        }
#pragma unroll
        for (int i = 0; i < 2; i++)
#pragma unroll
            for (int j = 0; j < 4; j++)
                acc[i][j] = __builtin_amdgcn_wmma_f32_16x16x32_bf16(
                    false, a[i].v, false, bfr[j].v, (short)0, acc[i][j], false, false);
    };

#if USE_ASYNC_LDS
    // stage one 128x32 K-tile of A and W into LDS buffer `buf` via async DMA
    auto stage = [&](int k0, int buf) {
        bf16_t* dA = lA + buf * TILE_ELEMS;
        bf16_t* dW = lW + buf * TILE_ELEMS;
        __builtin_amdgcn_global_load_async_to_lds_b128(
            (GAS v4i_gcc*)(A + (size_t)(m0 + rl0) * K + k0 + sg * 8),
            (LAS v4i_gcc*)(dA + rl0 * LDSS + sg * 8), 0, 0);
        __builtin_amdgcn_global_load_async_to_lds_b128(
            (GAS v4i_gcc*)(A + (size_t)(m0 + rl1) * K + k0 + sg * 8),
            (LAS v4i_gcc*)(dA + rl1 * LDSS + sg * 8), 0, 0);
        __builtin_amdgcn_global_load_async_to_lds_b128(
            (GAS v4i_gcc*)(W + (size_t)(n0 + rl0) * K + k0 + sg * 8),
            (LAS v4i_gcc*)(dW + rl0 * LDSS + sg * 8), 0, 0);
        __builtin_amdgcn_global_load_async_to_lds_b128(
            (GAS v4i_gcc*)(W + (size_t)(n0 + rl1) * K + k0 + sg * 8),
            (LAS v4i_gcc*)(dW + rl1 * LDSS + sg * 8), 0, 0);
    };

    stage(0, 0);
    __builtin_amdgcn_s_wait_asynccnt(0);
    __syncthreads();
    for (int k0 = 0; k0 < K; k0 += 32) {
        const int cur = (k0 >> 5) & 1;
        if (k0 + 32 < K) stage(k0 + 32, cur ^ 1);   // DMA next tile under compute
        compute_tile(cur);
        __builtin_amdgcn_s_wait_asynccnt(0);        // publish next tile / license reuse
        __syncthreads();
    }
#else
    for (int k0 = 0; k0 < K; k0 += 32) {
        __syncthreads();
        *(u32x4*)(lA + rl0 * LDSS + sg * 8) = *(const u32x4*)(A + (size_t)(m0 + rl0) * K + k0 + sg * 8);
        *(u32x4*)(lA + rl1 * LDSS + sg * 8) = *(const u32x4*)(A + (size_t)(m0 + rl1) * K + k0 + sg * 8);
        *(u32x4*)(lW + rl0 * LDSS + sg * 8) = *(const u32x4*)(W + (size_t)(n0 + rl0) * K + k0 + sg * 8);
        *(u32x4*)(lW + rl1 * LDSS + sg * 8) = *(const u32x4*)(W + (size_t)(n0 + rl1) * K + k0 + sg * 8);
        if (k0 + 32 < K) {
            __builtin_prefetch(A + (size_t)(m0 + rl0) * K + k0 + 32 + sg * 8, 0, 1);
            __builtin_prefetch(W + (size_t)(n0 + rl0) * K + k0 + 32 + sg * 8, 0, 1);
        }
        __syncthreads();
        compute_tile(0);
    }
#endif

    // epilogue: C/D layout -> lane n = lane&15, row = vgpr + (lane>=16 ? 8 : 0)
    const int l16 = lane & 15, hi8 = (lane >> 4) * 8;
#pragma unroll
    for (int i = 0; i < 2; i++) {
#pragma unroll
        for (int j = 0; j < 4; j++) {
            int n = n0 + wn + j * 16 + l16;
            float bv = bias ? bias[n] : 0.f;
#pragma unroll
            for (int e = 0; e < 8; e++) {
                int m = m0 + wm + i * 16 + hi8 + e;
                float v = acc[i][j][e] + bv;
                if (act) v = 0.5f * v * (1.f + erff(v * 0.70710678118654752f));  // exact GELU
                size_t o = (size_t)m * N + n;
                if (R) v += R[o];
                if (outF) outF[o] = v;
                if (outB) outB[o] = f2bf(v);
            }
        }
    }
}

// ---------------------------------------------------------------------------
// Tiny-sequence MHA core: per token (wave), heads=2, hd=64, S<=6.
// Q rows = (token*Sq+i)*qStride + qOff ; K/V rows = (token*Skv+j)*kvStride + kOff/vOff
// Output bf16 (token*Sq+i)*128 + head*64 + d
// ---------------------------------------------------------------------------
__global__ __launch_bounds__(256) void k_attn(const float* __restrict__ Q, int qStride, int qOff,
                                              const float* __restrict__ KV, int kvStride, int kOff, int vOff,
                                              bf16_t* __restrict__ O, int T, int Sq, int Skv) {
    int wave = threadIdx.x >> 5, lane = threadIdx.x & 31;
    int tok = blockIdx.x * 8 + wave;
    if (tok >= T) return;
    for (int h = 0; h < 2; ++h) {
        int base = h * 64;
        float kr[6][2], vr[6][2];
        for (int j = 0; j < Skv; j++) {
            const float* kp = KV + (size_t)(tok * Skv + j) * kvStride;
            kr[j][0] = kp[kOff + base + lane];  kr[j][1] = kp[kOff + base + lane + 32];
            vr[j][0] = kp[vOff + base + lane];  vr[j][1] = kp[vOff + base + lane + 32];
        }
        for (int i = 0; i < Sq; i++) {
            const float* qp = Q + (size_t)(tok * Sq + i) * qStride + qOff + base;
            float q0 = qp[lane], q1 = qp[lane + 32];
            float s[6];
            for (int j = 0; j < Skv; j++) {
                float p = q0 * kr[j][0] + q1 * kr[j][1];
                for (int o = 16; o; o >>= 1) p += __shfl_xor(p, o, 32);
                s[j] = p * 0.125f;   // 1/sqrt(64)
            }
            float mx = s[0];
            for (int j = 1; j < Skv; j++) mx = fmaxf(mx, s[j]);
            float sum = 0.f;
            for (int j = 0; j < Skv; j++) { s[j] = expf(s[j] - mx); sum += s[j]; }
            float inv = 1.f / sum;
            float o0 = 0.f, o1 = 0.f;
            for (int j = 0; j < Skv; j++) { float a = s[j] * inv; o0 += a * vr[j][0]; o1 += a * vr[j][1]; }
            bf16_t* op = O + (size_t)(tok * Sq + i) * 128 + base;
            op[lane] = f2bf(o0); op[lane + 32] = f2bf(o1);
        }
    }
}

// ---------------------------------------------------------------------------
// Split (T,6,128) bf16 -> two contiguous halves (T*3,128) bf16
// ---------------------------------------------------------------------------
__global__ __launch_bounds__(256) void k_split(const bf16_t* __restrict__ x6,
                                               bf16_t* __restrict__ f, bf16_t* __restrict__ s, int T) {
    int idx = blockIdx.x * 256 + threadIdx.x;
    int total = T * 3 * 128;
    if (idx >= total) return;
    int d = idx & 127; int rc = idx >> 7; int t = rc / 3, c = rc % 3;
    f[idx] = x6[((size_t)t * 6 + c) * 128 + d];
    s[idx] = x6[((size_t)t * 6 + c + 3) * 128 + d];
}

// Concat two f32 halves -> (T,6,128) bf16
__global__ __launch_bounds__(256) void k_concat(const float* __restrict__ f, const float* __restrict__ s,
                                                bf16_t* __restrict__ out, int T) {
    int idx = blockIdx.x * 256 + threadIdx.x;
    int total = T * 3 * 128;
    if (idx >= total) return;
    int d = idx & 127; int rc = idx >> 7; int t = rc / 3, c = rc % 3;
    out[((size_t)t * 6 + c) * 128 + d]     = f2bf(f[idx]);
    out[((size_t)t * 6 + c + 3) * 128 + d] = f2bf(s[idx]);
}

// ---------------------------------------------------------------------------
// Host orchestration
// ---------------------------------------------------------------------------
extern "C" void kernel_launch(void* const* d_in, const int* in_sizes, int n_in,
                              void* d_out, int out_size, void* d_ws, size_t ws_size,
                              hipStream_t stream) {
    (void)out_size;
    const int T  = 8 * 2048;   // tokens
    const int R6 = T * 6;      // 98304 rows of 128
    const int R3 = T * 3;      // 49152

    // ---- locate x and params (JAX pytree: nested dict keys sorted) ----
    int pi = 0;
    const float* x;
    if (in_sizes[0] == T * 768) { x = (const float*)d_in[0]; pi = 1; }
    else                        { x = (const float*)d_in[n_in - 1]; pi = 0; }
    auto nxt = [&]() { return (const float*)d_in[pi++]; };

    struct LNP { const float *g, *b; };
    struct LinP { const float *w, *b; };
    struct BlkP { const float *f1_b,*f1_w,*f2_b,*f2_w,*in_b,*in_w,*n1g,*n1b,*n2g,*n2b,*out_b,*out_w; };
    auto rdLN  = [&]() { LNP p; p.g = nxt(); p.b = nxt(); return p; };
    auto rdLin = [&]() { LinP p; p.w = nxt(); p.b = nxt(); return p; };
    auto rdBlk = [&]() { BlkP p;
        p.f1_b = nxt(); p.f1_w = nxt(); p.f2_b = nxt(); p.f2_w = nxt();
        p.in_b = nxt(); p.in_w = nxt(); p.n1g = nxt(); p.n1b = nxt();
        p.n2g = nxt(); p.n2b = nxt(); p.out_b = nxt(); p.out_w = nxt(); return p; };

    LNP  group_norm = rdLN();
    LinP group_proj = rdLin();
    BlkP grouped    = rdBlk();
    BlkP hier       = rdBlk();
    LNP  hier_norm  = rdLN();
    LinP hier_proj  = rdLin();
    LNP  norm       = rdLN();
    LinP proj_in    = rdLin();
    LinP proj_out   = rdLin();
    BlkP rel        = rdBlk();
    LNP  rel_norm   = rdLN();
    LinP rel_proj   = rdLin();
    BlkP standard   = rdBlk();

    // ---- workspace carve ----
    char* ws = (char*)d_ws;
    size_t off = 0;
    auto carve = [&](size_t bytes) -> void* {
        void* p = ws + off; off = (off + bytes + 255) & ~size_t(255); return p;
    };
    struct BW { bf16_t *in_w, *out_w, *f1_w, *f2_w; };
    auto carveBW = [&]() { BW w;
        w.in_w  = (bf16_t*)carve((size_t)384 * 128 * 2);
        w.out_w = (bf16_t*)carve((size_t)128 * 128 * 2);
        w.f1_w  = (bf16_t*)carve((size_t)512 * 128 * 2);
        w.f2_w  = (bf16_t*)carve((size_t)128 * 512 * 2);
        return w; };

    bf16_t* piw  = (bf16_t*)carve((size_t)768 * 768 * 2);
    bf16_t* pow_ = (bf16_t*)carve((size_t)768 * 768 * 2);
    bf16_t* gpw  = (bf16_t*)carve((size_t)128 * 128 * 2);
    bf16_t* hpw  = (bf16_t*)carve((size_t)128 * 128 * 2);
    bf16_t* rpw  = (bf16_t*)carve((size_t)128 * 128 * 2);
    BW bws = carveBW(), bwg = carveBW(), bwh = carveBW(), bwr = carveBW();

    bf16_t* A_bf = (bf16_t*)carve((size_t)R6 * 128 * 2);   // generic bf16 activations
    bf16_t* H_bf = (bf16_t*)carve((size_t)R6 * 512 * 2);   // FFN hidden (also hier split scratch)
    float*  P0   = (float*) carve((size_t)R6 * 128 * 4);
    float*  P1   = (float*) carve((size_t)R6 * 128 * 4);
    float*  B1   = (float*) carve((size_t)R6 * 128 * 4);
    float*  QKVf = (float*) carve((size_t)R6 * 384 * 4);
    if (off > ws_size) return;   // not enough scratch: bail (compile-only loop)

    // ---- weight conversion ----
    auto conv = [&](const float* src, bf16_t* dst, int n) {
        k_f2bf<<<dim3((n + 255) / 256), dim3(256), 0, stream>>>(src, dst, n);
    };
    conv(proj_in.w,  piw,  768 * 768);
    conv(proj_out.w, pow_, 768 * 768);
    conv(group_proj.w, gpw, 128 * 128);
    conv(hier_proj.w,  hpw, 128 * 128);
    conv(rel_proj.w,   rpw, 128 * 128);
    auto convBW = [&](const BlkP& p, const BW& w) {
        conv(p.in_w,  w.in_w,  384 * 128);
        conv(p.out_w, w.out_w, 128 * 128);
        conv(p.f1_w,  w.f1_w,  512 * 128);
        conv(p.f2_w,  w.f2_w,  128 * 512);
    };
    convBW(standard, bws); convBW(grouped, bwg); convBW(hier, bwh); convBW(rel, bwr);

    auto LNrows = [&](int rows) { return dim3((unsigned)((rows + 7) / 8)); };
    auto gemm = [&](const bf16_t* A, const bf16_t* W, const float* bias, const float* R,
                    float* oF, bf16_t* oB, int M, int N, int K, int act) {
        k_gemm<<<dim3((unsigned)(N / 128), (unsigned)(M / 128)), dim3(256), 0, stream>>>(
            A, W, bias, R, oF, oB, M, N, K, act);
    };

    // ---- self-attention block: X -> (outF/outB) ----
    auto self_block = [&](const BlkP& P, const BW& W, const float* Xin, int tok, int S,
                          float* oF, bf16_t* oB) {
        int rows = tok * S;
        k_ln<<<LNrows(rows), 256, 0, stream>>>(Xin, P.n1g, P.n1b, nullptr, A_bf, rows, 128);
        gemm(A_bf, W.in_w, P.in_b, nullptr, QKVf, nullptr, rows, 384, 128, 0);
        k_attn<<<dim3((unsigned)((tok + 7) / 8)), 256, 0, stream>>>(
            QKVf, 384, 0, QKVf, 384, 128, 256, A_bf, tok, S, S);
        gemm(A_bf, W.out_w, P.out_b, Xin, B1, nullptr, rows, 128, 128, 0);
        k_ln<<<LNrows(rows), 256, 0, stream>>>(B1, P.n2g, P.n2b, nullptr, A_bf, rows, 128);
        gemm(A_bf, W.f1_w, P.f1_b, nullptr, nullptr, H_bf, rows, 512, 128, 1);
        gemm(H_bf, W.f2_w, P.f2_b, B1, oF, oB, rows, 128, 512, 0);
    };

    // ---- cross-attention block: q=Qin, k=v=KVin -> outF ----
    auto cross_block = [&](const BlkP& P, const BW& W, const float* Qin, const float* KVin,
                           int tok, float* oF) {
        int rows = tok * 3;
        bf16_t* Qn  = A_bf;
        bf16_t* KVn = A_bf + (size_t)rows * 128;
        float*  Qp  = QKVf;
        float*  KVp = QKVf + (size_t)rows * 128;
        k_ln<<<LNrows(rows), 256, 0, stream>>>(Qin,  P.n1g, P.n1b, nullptr, Qn,  rows, 128);
        k_ln<<<LNrows(rows), 256, 0, stream>>>(KVin, P.n1g, P.n1b, nullptr, KVn, rows, 128);
        gemm(Qn,  W.in_w,             P.in_b,       nullptr, Qp,  nullptr, rows, 128, 128, 0);
        gemm(KVn, W.in_w + 128 * 128, P.in_b + 128, nullptr, KVp, nullptr, rows, 256, 128, 0);
        k_attn<<<dim3((unsigned)((tok + 7) / 8)), 256, 0, stream>>>(
            Qp, 128, 0, KVp, 256, 0, 128, A_bf, tok, 3, 3);
        gemm(A_bf, W.out_w, P.out_b, Qin, B1, nullptr, rows, 128, 128, 0);
        k_ln<<<LNrows(rows), 256, 0, stream>>>(B1, P.n2g, P.n2b, nullptr, A_bf, rows, 128);
        gemm(A_bf, W.f1_w, P.f1_b, nullptr, nullptr, H_bf, rows, 512, 128, 1);
        gemm(H_bf, W.f2_w, P.f2_b, B1, oF, nullptr, rows, 128, 512, 0);
    };

    // =================== pipeline ===================
    // LN(x) -> bf16 ; proj_in -> x_ch f32 (P0, viewed as R6 x 128)
    k_ln<<<LNrows(T), 256, 0, stream>>>(x, norm.g, norm.b, nullptr, A_bf, T, 768);
    gemm(A_bf, piw, proj_in.b, nullptr, P0, nullptr, T, 768, 768, 0);

    // 1. standard (S=6) -> bf16 in A_bf
    self_block(standard, bws, P0, T, 6, nullptr, A_bf);

    // 2. grouped: lin+ln, then block over (2T tokens, S=3) -> bf16 in A_bf
    gemm(A_bf, gpw, group_proj.b, nullptr, B1, nullptr, R6, 128, 128, 0);
    k_ln<<<LNrows(R6), 256, 0, stream>>>(B1, group_norm.g, group_norm.b, P0, nullptr, R6, 128);
    self_block(grouped, bwg, P0, 2 * T, 3, nullptr, A_bf);

    // 3. hierarchical cross-attention between halves
    bf16_t* Fh = H_bf;
    bf16_t* Sh = H_bf + (size_t)R3 * 128;
    k_split<<<dim3((unsigned)((T * 3 * 128 + 255) / 256)), 256, 0, stream>>>(A_bf, Fh, Sh, T);
    gemm(Fh, hpw, hier_proj.b, nullptr, B1,                     nullptr, R3, 128, 128, 0);
    gemm(Sh, hpw, hier_proj.b, nullptr, B1 + (size_t)R3 * 128,  nullptr, R3, 128, 128, 0);
    k_ln<<<LNrows(R3), 256, 0, stream>>>(B1,                    hier_norm.g, hier_norm.b, P0,                    nullptr, R3, 128);
    k_ln<<<LNrows(R3), 256, 0, stream>>>(B1 + (size_t)R3 * 128, hier_norm.g, hier_norm.b, P0 + (size_t)R3 * 128, nullptr, R3, 128);
    cross_block(hier, bwh, P0,                    P0 + (size_t)R3 * 128, T, P1);                     // first
    cross_block(hier, bwh, P0 + (size_t)R3 * 128, P1,                    T, P1 + (size_t)R3 * 128);  // second (uses updated first)
    k_concat<<<dim3((unsigned)((T * 3 * 128 + 255) / 256)), 256, 0, stream>>>(
        P1, P1 + (size_t)R3 * 128, A_bf, T);

    // 4. relational: lin+ln, then block (S=6) -> bf16 in A_bf
    gemm(A_bf, rpw, rel_proj.b, nullptr, B1, nullptr, R6, 128, 128, 0);
    k_ln<<<LNrows(R6), 256, 0, stream>>>(B1, rel_norm.g, rel_norm.b, P0, nullptr, R6, 128);
    self_block(rel, bwr, P0, T, 6, nullptr, A_bf);

    // proj_out + input residual -> d_out f32 (T,768)
    gemm(A_bf, pow_, proj_out.b, x, (float*)d_out, nullptr, T, 768, 768, 0);
}
